// SelfAttnHead_8821862826452
// MI455X (gfx1250) — matrix operand order repaired and checked by reading, not compiled, not run on hardware
//
#include <hip/hip_runtime.h>
#include <hip/hip_bf16.h>
#include <math.h>

// Sizes from the reference
constexpr int Bsz = 8;
constexpr int Tsz = 2048;
constexpr int Esz = 1024;
constexpr int Hsz = 64;

typedef _Float16 f16;
typedef __attribute__((ext_vector_type(16))) _Float16 v16h;
typedef __attribute__((ext_vector_type(8)))  _Float16 v8h;
typedef __attribute__((ext_vector_type(8)))  float    v8f;
typedef __attribute__((ext_vector_type(4)))  float    v4f;

__device__ __forceinline__ v8f wmma_f16(v16h a, v16h b, v8f c) {
  // D = A(16x32 f16) * B(32x16 f16) + C(16x16 f32)
  return __builtin_amdgcn_wmma_f32_16x16x32_f16(
      false, a, false, b, (short)0, c, false, false);
}

// Build a v16h A-operand from two contiguous 8-half chunks.
// A 16x32 layout (contiguous k order): lane row = lane%16, kbase = (lane/16)*16,
//   a[0..15] = K = kbase+i
__device__ __forceinline__ v16h make_a(const f16* lo, const f16* hi) {
  union { v16h v; v8h h[2]; } u;
  u.h[0] = *(const v8h*)lo;
  u.h[1] = *(const v8h*)hi;
  return u.v;
}

// ---- CDNA5 async global->LDS copy (ASYNCcnt-tracked) ----------------------
// Per-lane: LDS[lds_off] <= MEM[gaddr], 16 bytes. One instruction per wave.
__device__ __forceinline__ void async_b128(unsigned lds_off,
                                           unsigned long long gaddr) {
  asm volatile("global_load_async_to_lds_b128 %0, %1, off"
               :: "v"(lds_off), "v"(gaddr) : "memory");
}
// wait until <= N async ops outstanding (loads complete in order)
__device__ __forceinline__ void wait_async_le8() {
  asm volatile("s_wait_asynccnt 0x8" ::: "memory");
}
__device__ __forceinline__ void wait_async_0() {
  asm volatile("s_wait_asynccnt 0x0" ::: "memory");
}
// low 32 bits of a generic pointer to LDS == LDS byte offset (aperture rule)
__device__ __forceinline__ unsigned lds_off32(const void* p) {
  return (unsigned)(unsigned long long)p;
}

// ---------------------------------------------------------------------------
// Kernel 1: Q = x@Wq + bq (row-major f16), V = x@Wv + bv stored TRANSPOSED
// as v16T[(b*H + d)*T + t] so the attention kernel can stage V tiles with
// plain contiguous async copies. (Keys reuse Q per the reference's bug.)
// 256 threads = 8 waves: waves 0-3 -> Q, waves 4-7 -> V, same 64-row tile.
// ---------------------------------------------------------------------------
__global__ __launch_bounds__(256) void proj_kernel(
    const float* __restrict__ x,
    const float* __restrict__ Wq, const float* __restrict__ bq,
    const float* __restrict__ Wv, const float* __restrict__ bv,
    f16* __restrict__ q16, f16* __restrict__ v16T) {
  __shared__ alignas(32) f16 xs[64 * 32];    // x chunk, row-major [row][k]
  __shared__ alignas(32) f16 wqT[64 * 32];   // Wq chunk transposed [n][k]
  __shared__ alignas(32) f16 wvT[64 * 32];   // Wv chunk transposed [n][k]

  const int tid  = threadIdx.x;
  const int lane = tid & 31;
  const int wave = tid >> 5;
  const int row0 = blockIdx.x * 64;          // base row in (B*T, H)
  const bool isV = wave >= 4;
  const int wrow = (wave & 3) * 16;

  v8f acc[4] = {{}, {}, {}, {}};

  const int arow  = wrow + (lane & 15);
  const int kbo   = (lane >> 4) * 8;
  const int kb2   = (lane >> 4) * 16;
  const int nlane = lane & 15;

  for (int k0 = 0; k0 < Esz; k0 += 32) {
    __syncthreads();
    {  // stage x chunk: 64 rows x 32 cols, f32 -> f16
      int r = tid >> 2;
      int c = (tid & 3) * 8;
      const float* src = x + (size_t)(row0 + r) * Esz + k0 + c;
      v4f x0 = *(const v4f*)(src);
      v4f x1 = *(const v4f*)(src + 4);
      v8h h;
      #pragma unroll
      for (int j = 0; j < 4; ++j) { h[j] = (f16)x0[j]; h[4 + j] = (f16)x1[j]; }
      *(v8h*)&xs[r * 32 + c] = h;
    }
    {  // stage Wq/Wv chunk transposed: wT[n*32+k] = W[(k0+k)*H + n]
      int k  = tid >> 3;
      int n0 = (tid & 7) * 8;
      const float* sq = Wq + (size_t)(k0 + k) * Hsz + n0;
      const float* sv = Wv + (size_t)(k0 + k) * Hsz + n0;
      #pragma unroll
      for (int j = 0; j < 8; ++j) {
        wqT[(n0 + j) * 32 + k] = (f16)sq[j];
        wvT[(n0 + j) * 32 + k] = (f16)sv[j];
      }
    }
    __syncthreads();

    v16h a = make_a(&xs[arow * 32 + kb2], &xs[arow * 32 + kb2 + 8]);
    const f16* wT = isV ? wvT : wqT;
    #pragma unroll
    for (int n = 0; n < 4; ++n) {
      v16h b = *(const v16h*)&wT[(n * 16 + nlane) * 32 + kb2];
      acc[n] = wmma_f16(a, b, acc[n]);
    }
  }

  const int rhalf = (lane >> 4) * 8;
  if (!isV) {
    #pragma unroll
    for (int n = 0; n < 4; ++n) {
      int col = n * 16 + nlane;
      float bcol = bq[col];
      #pragma unroll
      for (int r = 0; r < 8; ++r) {
        int grow = row0 + wrow + rhalf + r;
        q16[(size_t)grow * Hsz + col] = (f16)(acc[n][r] + bcol);
      }
    }
  } else {
    const int bIdx = row0 / Tsz;
    const int t0   = (row0 % Tsz) + wrow + rhalf;  // multiple of 8 -> 16B align
    #pragma unroll
    for (int n = 0; n < 4; ++n) {
      int col = n * 16 + nlane;
      float bcol = bv[col];
      v8h h;
      #pragma unroll
      for (int r = 0; r < 8; ++r) h[r] = (f16)(acc[n][r] + bcol);
      *(v8h*)&v16T[((size_t)bIdx * Hsz + col) * Tsz + t0] = h;
    }
  }
}

// ---------------------------------------------------------------------------
// Kernel 2: flash attention, 64-key tiles, double-buffered async LDS staging.
// 128 threads = 4 waves; wave w owns queries [blk*64 + w*16, +16).
// ---------------------------------------------------------------------------
__global__ __launch_bounds__(128) void attn_kernel(
    const f16* __restrict__ q16,   // also the keys (reference reuses q_proj)
    const f16* __restrict__ v16T,  // V transposed: [(b*H + d)*T + t]
    const int* __restrict__ mask,  // (B, T)
    float* __restrict__ out) {     // (B, T, H) f32
  __shared__ alignas(32) f16 Ks[2][64 * 64];   // K tile row-major [key][d]
  __shared__ alignas(32) f16 Vs[2][64 * 64];   // V tile [d][key]
  __shared__ alignas(32) f16 Ps[4][16 * 64];   // per-wave P scratch [q][key]

  const int tid  = threadIdx.x;
  const int lane = tid & 31;
  const int wave = tid >> 5;
  const int b    = blockIdx.y;
  const int qblk = blockIdx.x;
  const int q0w  = qblk * 64 + wave * 16;
  const size_t rowbase = (size_t)b * Tsz;

  const int nlane = lane & 15;
  const int rhalf = (lane >> 4) * 8;
  const int kbo   = (lane >> 4) * 8;
  const int kb2   = (lane >> 4) * 16;

  // --- async tile staging: 8 issues/wave per tile (4 K + 4 V), 16B/lane ---
  const f16* vbat = v16T + (size_t)b * Hsz * Tsz;  // + d*T + t
  auto stage = [&](int kb, int buf) {
    const f16* ksrc = q16 + (rowbase + kb * 64) * Hsz + tid * 8;  // contiguous
    unsigned kdst = lds_off32(&Ks[buf][0]) + tid * 16;
    #pragma unroll
    for (int i = 0; i < 4; ++i)
      async_b128(kdst + i * 2048,
                 (unsigned long long)(ksrc + i * 1024));
    const int vrow = tid >> 3;          // 0..15
    const int vcol = (tid & 7) * 8;     // halves within a 64-key row
    const f16* vsrc = vbat + kb * 64 + vcol;
    unsigned vdst = lds_off32(&Vs[buf][0]) + vrow * 128 + vcol * 2;
    #pragma unroll
    for (int i = 0; i < 4; ++i)
      async_b128(vdst + i * 16 * 128,
                 (unsigned long long)(vsrc + (size_t)(vrow + i * 16) * Tsz));
  };

  // Q tile -> A-layout registers for the whole kernel
  const int arow = q0w + nlane;
  v16h aq[2];
  #pragma unroll
  for (int c = 0; c < 2; ++c) {
    const f16* src = q16 + (rowbase + arow) * Hsz + c * 32 + kb2;
    aq[c] = make_a(src, src + 8);
  }

  v8f o[4] = {{}, {}, {}, {}};
  float m[8], l[8];
  #pragma unroll
  for (int r = 0; r < 8; ++r) { m[r] = -INFINITY; l[r] = 0.f; }

  const int nkb = qblk + 1;                       // causal: 64-key tiles
  const float scale2 = 0.125f * 1.44269504088896340736f;  // (1/sqrt(H))*log2(e)

  stage(0, 0);

  for (int kb = 0; kb < nkb; ++kb) {
    const int cur = kb & 1;
    if (kb + 1 < nkb) { stage(kb + 1, cur ^ 1); wait_async_le8(); }
    else              { wait_async_0(); }
    __syncthreads();                              // tile kb fully in LDS

    int mk[4];
    const int key0 = kb * 64 + nlane;
    #pragma unroll
    for (int g = 0; g < 4; ++g) mk[g] = mask[rowbase + key0 + g * 16];

    // S = Q K^T : four 16-key column groups, reduction in 2 chunks of 32
    v8f sg[4] = {{}, {}, {}, {}};
    #pragma unroll
    for (int c = 0; c < 2; ++c) {
      #pragma unroll
      for (int g = 0; g < 4; ++g) {
        v16h bk = *(const v16h*)&Ks[cur][(g * 16 + nlane) * 64 + c * 32 + kb2];
        sg[g] = wmma_f16(aq[c], bk, sg[g]);
      }
    }

    // online softmax (base-2), per accumulator row slot r
    f16* P = &Ps[wave][0];
    #pragma unroll
    for (int r = 0; r < 8; ++r) {
      int qrow = q0w + rhalf + r;
      float vg[4];
      float tmax = -INFINITY;
      #pragma unroll
      for (int g = 0; g < 4; ++g) {
        bool al = (key0 + g * 16 <= qrow) && (mk[g] != 0);
        vg[g] = al ? sg[g][r] * scale2 : -INFINITY;
        tmax = fmaxf(tmax, vg[g]);
      }
      #pragma unroll
      for (int sh = 1; sh < 16; sh <<= 1)
        tmax = fmaxf(tmax, __shfl_xor(tmax, sh, 32));
      float mn = fmaxf(m[r], tmax);
      float alpha = (mn == -INFINITY) ? 1.0f : exp2f(m[r] - mn);
      float ps = 0.f;
      f16 ph[4];
      #pragma unroll
      for (int g = 0; g < 4; ++g) {
        float p = (vg[g] == -INFINITY) ? 0.f : exp2f(vg[g] - mn);
        ps += p;
        ph[g] = (f16)p;
      }
      #pragma unroll
      for (int sh = 1; sh < 16; sh <<= 1)
        ps += __shfl_xor(ps, sh, 32);
      l[r] = l[r] * alpha + ps;
      m[r] = mn;
      #pragma unroll
      for (int n2 = 0; n2 < 4; ++n2) o[n2][r] *= alpha;
      #pragma unroll
      for (int g = 0; g < 4; ++g)
        P[(rhalf + r) * 64 + g * 16 + nlane] = ph[g];
    }

    // P (C layout) -> per-wave LDS -> A layout (two K-chunks of 32)
    const f16* prow = &P[nlane * 64];
    v16h ap0 = make_a(prow + kb2,      prow + kb2 + 8);
    v16h ap1 = make_a(prow + 32 + kb2, prow + 32 + kb2 + 8);

    // O += P * V  (B operand contiguous from [d][key] V tile)
    #pragma unroll
    for (int n2 = 0; n2 < 4; ++n2) {
      const f16* vr = &Vs[cur][(n2 * 16 + nlane) * 64];
      o[n2] = wmma_f16(ap0, *(const v16h*)(vr + kb2),      o[n2]);
      o[n2] = wmma_f16(ap1, *(const v16h*)(vr + 32 + kb2), o[n2]);
    }

    __syncthreads();   // all waves done reading tile kb before it is restaged
  }

  // epilogue: normalize; fully-masked rows emit v[q] (reference's diag fix)
  #pragma unroll
  for (int r = 0; r < 8; ++r) {
    int qrow = q0w + rhalf + r;
    float lr = l[r];
    float inv = (lr > 0.f) ? 1.0f / lr : 0.f;
    #pragma unroll
    for (int n2 = 0; n2 < 4; ++n2) {
      int col = n2 * 16 + nlane;
      float val = (lr > 0.f)
                      ? o[n2][r] * inv
                      : (float)vbat[(size_t)col * Tsz + qrow];
      out[(rowbase + qrow) * Hsz + col] = val;
    }
  }
}

// ---------------------------------------------------------------------------
extern "C" void kernel_launch(void* const* d_in, const int* in_sizes, int n_in,
                              void* d_out, int out_size, void* d_ws,
                              size_t ws_size, hipStream_t stream) {
  const float* x  = (const float*)d_in[0];
  const float* Wq = (const float*)d_in[1];
  const float* bq = (const float*)d_in[2];
  const float* Wv = (const float*)d_in[3];
  const float* bv = (const float*)d_in[4];
  const int* msk  = (const int*)d_in[5];
  float* out = (float*)d_out;

  // workspace: q16 row-major (2MB) | v16 transposed (2MB)
  f16* q16  = (f16*)d_ws;
  f16* v16T = q16 + (size_t)Bsz * Tsz * Hsz;

  proj_kernel<<<(Bsz * Tsz) / 64, 256, 0, stream>>>(x, Wq, bq, Wv, bv,
                                                    q16, v16T);
  attn_kernel<<<dim3(Tsz / 64, Bsz), 128, 0, stream>>>(q16, v16T, msk, out);
}
